// MultiHeadAttention_70712341561681
// MI455X (gfx1250) — matrix operand order repaired and checked by reading, not compiled, not run on hardware
//
#include <hip/hip_runtime.h>

typedef __attribute__((ext_vector_type(16))) _Float16 v16h;
typedef __attribute__((ext_vector_type(8)))  _Float16 h8;
typedef __attribute__((ext_vector_type(4)))  _Float16 h4;
typedef __attribute__((ext_vector_type(8)))  float    v8f;
typedef __attribute__((ext_vector_type(4)))  float    f4;
typedef __attribute__((ext_vector_type(4)))  int      i4;
// Address-space-qualified int4 for the CDNA5 async LDS->global builtin:
typedef __attribute__((address_space(1))) i4 gas_i4;   // global
typedef __attribute__((address_space(3))) i4 las_i4;   // LDS

// Problem dims (fixed by reference)
static constexpr int cB  = 2;
static constexpr int cT  = 2048;
static constexpr int cC  = 1024;
static constexpr int cH  = 16;
static constexpr int cHD = 64;

#if __has_builtin(__builtin_amdgcn_global_store_async_from_lds_b128)
#define HAVE_ASYNC_STORE 1
#else
#define HAVE_ASYNC_STORE 0
#endif

// ---------- fragment loaders (CDNA5 WMMA 16x16x32 f16 layouts) ----------
// A-matrix 16x32: lane L (0-15) = row M=L, VGPR0-3 K=0..7, VGPR4-7 K=16..23;
// lanes 16-31 same rows, K=8..15 and 24..31.  => chunks at d+hi*8 and d+16+hi*8.
__device__ __forceinline__ v16h load_a_frag(const _Float16* rowp, int dbase, int hi) {
  h8 lo = *(const h8*)(rowp + dbase + hi * 8);
  h8 hv = *(const h8*)(rowp + dbase + 16 + hi * 8);
  return __builtin_shufflevector(lo, hv, 0,1,2,3,4,5,6,7,8,9,10,11,12,13,14,15);
}
// B-matrix 32x16: lane n (0-15) = column N=n, K=0..7 and 16..23;
// lanes 16-31 K=8..15 and 24..31.  => chunks at d+hi*8 and d+16+hi*8.
__device__ __forceinline__ v16h load_b_frag(const _Float16* colp, int dbase, int hi) {
  h8 lo = *(const h8*)(colp + dbase + hi * 8);
  h8 hv = *(const h8*)(colp + dbase + 16 + hi * 8);
  return __builtin_shufflevector(lo, hv, 0,1,2,3,4,5,6,7,8,9,10,11,12,13,14,15);
}

// ---------- fp32 -> f16 conversion (vectorized x4) ----------
__global__ __launch_bounds__(256) void cvt_f32_f16(const float* __restrict__ in,
                                                   _Float16* __restrict__ out, int n4) {
  int i = blockIdx.x * 256 + threadIdx.x;
  if (i >= n4) return;
  f4 v = *(const f4*)(in + (size_t)i * 4);
  h4 o; o[0] = (_Float16)v[0]; o[1] = (_Float16)v[1];
  o[2] = (_Float16)v[2]; o[3] = (_Float16)v[3];
  *(h4*)(out + (size_t)i * 4) = o;
}

// ---------- WMMA GEMM:  C[M,N] = A[M,K] * W[N,K]^T (+bias) ----------
// One wave computes a 16x64 strip (4 accumulators reuse the A fragment).
__global__ __launch_bounds__(256) void gemm_f16_wmma(
    const _Float16* __restrict__ A, const _Float16* __restrict__ W,
    const float* __restrict__ bias, float* __restrict__ Cf,
    _Float16* __restrict__ Ch, int M, int N, int K) {
  const int wave = (blockIdx.x * 256 + threadIdx.x) >> 5;
  const int lane = threadIdx.x & 31;
  const int nt4  = N >> 6;
  const int mt   = wave / nt4;
  const int nt   = wave % nt4;
  if (mt * 16 >= M) return;
  const int hi = lane >> 4;
  const int lr = lane & 15;
  const int n0 = nt * 64;
  const _Float16* arow = A + ((size_t)(mt * 16 + lr)) * K;
  const _Float16* wrow = W + ((size_t)(n0 + lr)) * K;
  v8f c0 = {}, c1 = {}, c2 = {}, c3 = {};
  for (int kb = 0; kb < K; kb += 32) {
    v16h a  = load_a_frag(arow, kb, hi);
    v16h b0 = load_b_frag(wrow,                  kb, hi);
    v16h b1 = load_b_frag(wrow + 16 * (size_t)K, kb, hi);
    v16h b2 = load_b_frag(wrow + 32 * (size_t)K, kb, hi);
    v16h b3 = load_b_frag(wrow + 48 * (size_t)K, kb, hi);
    c0 = __builtin_amdgcn_wmma_f32_16x16x32_f16(false, a, false, b0, (short)0, c0, false, false);
    c1 = __builtin_amdgcn_wmma_f32_16x16x32_f16(false, a, false, b1, (short)0, c1, false, false);
    c2 = __builtin_amdgcn_wmma_f32_16x16x32_f16(false, a, false, b2, (short)0, c2, false, false);
    c3 = __builtin_amdgcn_wmma_f32_16x16x32_f16(false, a, false, b3, (short)0, c3, false, false);
  }
  // D layout: VGPR r, lanes<16 -> M=r, lanes>=16 -> M=r+8; N = lane&15.
  #pragma unroll
  for (int r = 0; r < 8; ++r) {
    const size_t orow = (size_t)(mt * 16 + r + hi * 8);
    float vals[4] = { c0[r], c1[r], c2[r], c3[r] };
    #pragma unroll
    for (int t = 0; t < 4; ++t) {
      const int col = n0 + t * 16 + lr;
      if (Cf) Cf[orow * N + col] = vals[t] + (bias ? bias[col] : 0.0f);
      else    Ch[orow * N + col] = (_Float16)vals[t];
    }
  }
}

// ---------- fused ALiBi logits + causal mask + softmax ----------
// One block owns 16 key rows x 2048 query cols; logits stripe lives in LDS
// (16*2052*4B ~ 128 KB, fits the 320 KB CDNA5 WGP LDS); weights written once.
__global__ __launch_bounds__(256) void attn_softmax(
    const _Float16* __restrict__ q16, const _Float16* __restrict__ k16,
    float* __restrict__ wout) {
  __shared__ float lds[16][2052];   // +4 pad kills stride-8-row bank conflicts
  const int s  = blockIdx.x & 127;        // stripe index (T/16 = 128)
  const int bh = blockIdx.x >> 7;         // b*H + h
  const int h  = bh & 15;
  const int b  = bh >> 4;
  const int T0 = s * 16;
  const int lane = threadIdx.x & 31;
  const int wid  = threadIdx.x >> 5;
  const int hi   = lane >> 4;
  const int lr   = lane & 15;
  const float slope = exp2f(-0.5f * (float)h);   // 2^(h * -8/H), H=16

  // A fragment: 16 key rows [T0, T0+16), contraction over HD=64 (two K=32 steps)
  const _Float16* krow = k16 + ((size_t)(b * cT + T0 + lr)) * cC + h * cHD;
  const v16h a0 = load_a_frag(krow, 0, hi);
  const v16h a1 = load_a_frag(krow, 32, hi);

  // Pass 1: WMMA logit tiles (+ALiBi bias, causal mask) into LDS
  const int nTiles = s + 1;               // causal: only tiles with t0 <= T0
  for (int j = wid; j < nTiles; j += 8) {
    const int t0 = j * 16;
    const _Float16* qrow = q16 + ((size_t)(b * cT + t0 + lr)) * cC + h * cHD;
    v16h b0 = load_b_frag(qrow, 0, hi);
    v16h b1 = load_b_frag(qrow, 32, hi);
    v8f c = {};
    c = __builtin_amdgcn_wmma_f32_16x16x32_f16(false, a0, false, b0, (short)0, c, false, false);
    c = __builtin_amdgcn_wmma_f32_16x16x32_f16(false, a1, false, b1, (short)0, c, false, false);
    #pragma unroll
    for (int r = 0; r < 8; ++r) {
      const int m  = r + hi * 8;
      const int Tg = T0 + m;
      const int tg = t0 + lr;
      float v = c[r];
      v = (Tg >= tg) ? (v - slope * (float)(Tg - tg)) : -__builtin_inff();
      lds[m][t0 + lr] = v;
    }
  }
  __syncthreads();

  // Pass 2: per-row softmax, normalized in place (one half-wave per row).
  // Stores exp(l-m) (or 0 for the masked tail) back to LDS, then scales by
  // 1/sum with the same 16 threads -- no extra barrier, exp evaluated once.
  {
    const int rowi = threadIdx.x >> 4;
    const int g    = threadIdx.x & 15;
    const int lim  = T0 + rowi + 1;       // valid cols: [0, Tg]
    float m = -__builtin_inff();
    for (int cix = g; cix < lim; cix += 16) m = fmaxf(m, lds[rowi][cix]);
    #pragma unroll
    for (int o = 8; o; o >>= 1) m = fmaxf(m, __shfl_xor(m, o, 32));
    float sum = 0.0f;
    for (int cix = g; cix < cT; cix += 16) {
      const float e = (cix < lim) ? __expf(lds[rowi][cix] - m) : 0.0f;
      sum += e;
      lds[rowi][cix] = e;                 // also zero-fills the masked tail
    }
    #pragma unroll
    for (int o = 8; o; o >>= 1) sum += __shfl_xor(sum, o, 32);
    const float inv = 1.0f / sum;
    for (int cix = g; cix < lim; cix += 16) lds[rowi][cix] *= inv;
  }
  __syncthreads();

  // Pass 3: stream the fully-normalized 16x2048 stripe to HBM.
  float* wbase = wout + ((size_t)bh * cT + T0) * cT;
#if HAVE_ASYNC_STORE
  // CDNA5 async LDS->global DMA path (ASYNCcnt-tracked, b128 per lane).
  for (int v4 = threadIdx.x; v4 < 16 * (cT / 4); v4 += 256) {
    const int rowi = v4 >> 9;             // cT/4 = 512 float4 per row
    const int c4   = (v4 & 511) << 2;
    float* gp = wbase + (size_t)rowi * cT + c4;
    __builtin_amdgcn_global_store_async_from_lds_b128(
        (gas_i4*)gp, (las_i4*)&lds[rowi][c4], 0, 0);
  }
#if __has_builtin(__builtin_amdgcn_s_wait_asynccnt)
  __builtin_amdgcn_s_wait_asynccnt(0);
#else
  asm volatile("s_wait_asynccnt 0" ::: "memory");
#endif
#else
  // Fallback: nontemporal float4 register stores (537 MB once-written stream;
  // keep it out of L2).
  for (int v4 = threadIdx.x; v4 < 16 * (cT / 4); v4 += 256) {
    const int rowi = v4 >> 9;
    const int c4   = (v4 & 511) << 2;
    const f4 o = *(const f4*)&lds[rowi][c4];
    __builtin_nontemporal_store(o, (f4*)(wbase + (size_t)rowi * cT + c4));
  }
#endif
}

// ---------- att = diag(weights) * v  (f16 out for the final GEMM) ----------
__global__ __launch_bounds__(256) void diag_scale(
    const float* __restrict__ wts, const _Float16* __restrict__ v16,
    _Float16* __restrict__ att16) {
  const size_t idx = (size_t)blockIdx.x * 256 + threadIdx.x;
  const int    c   = (int)(idx & (cC - 1));
  const size_t bt  = idx >> 10;
  const int    t   = (int)(bt & (cT - 1));
  const int    b   = (int)(bt >> 11);
  const int    h   = c >> 6;
  const float  d   = wts[((size_t)(b * cH + h) * cT + t) * cT + t];
  att16[idx] = (_Float16)(d * (float)v16[idx]);
}

extern "C" void kernel_launch(void* const* d_in, const int* in_sizes, int n_in,
                              void* d_out, int out_size, void* d_ws, size_t ws_size,
                              hipStream_t stream) {
  (void)in_sizes; (void)n_in; (void)out_size; (void)ws_size;
  const float* x  = (const float*)d_in[0];
  const float* Wq = (const float*)d_in[1];
  const float* Wk = (const float*)d_in[2];
  const float* Wv = (const float*)d_in[3];
  const float* Wp = (const float*)d_in[4];
  const float* bp = (const float*)d_in[5];

  float* out = (float*)d_out;                               // [B,T,C]
  float* wts = out + (size_t)cB * cT * cC;                  // [B,H,T,T]

  const size_t BT_C = (size_t)cB * cT * cC;                 // 4,194,304
  const size_t W_N  = (size_t)cC * cC;                      // 1,048,576
  char* p = (char*)d_ws;
  _Float16* x16   = (_Float16*)p; p += BT_C * 2;
  _Float16* wq16  = (_Float16*)p; p += W_N * 2;
  _Float16* wk16  = (_Float16*)p; p += W_N * 2;
  _Float16* wv16  = (_Float16*)p; p += W_N * 2;
  _Float16* wp16  = (_Float16*)p; p += W_N * 2;
  _Float16* q16   = (_Float16*)p; p += BT_C * 2;
  _Float16* k16   = (_Float16*)p; p += BT_C * 2;
  _Float16* v16   = (_Float16*)p; p += BT_C * 2;
  _Float16* att16 = (_Float16*)p; p += BT_C * 2;

  // fp32 -> f16
  cvt_f32_f16<<<(int)(BT_C / 4 / 256), 256, 0, stream>>>(x, x16, (int)(BT_C / 4));
  cvt_f32_f16<<<(int)(W_N / 4 / 256), 256, 0, stream>>>(Wq, wq16, (int)(W_N / 4));
  cvt_f32_f16<<<(int)(W_N / 4 / 256), 256, 0, stream>>>(Wk, wk16, (int)(W_N / 4));
  cvt_f32_f16<<<(int)(W_N / 4 / 256), 256, 0, stream>>>(Wv, wv16, (int)(W_N / 4));
  cvt_f32_f16<<<(int)(W_N / 4 / 256), 256, 0, stream>>>(Wp, wp16, (int)(W_N / 4));

  // projections: [4096,1024] @ [1024,1024]^T -> f16 workspace
  const int M = cB * cT, N = cC, K = cC;
  const int gemmBlocks = (M / 16) * (N / 64) / 8;           // 512 blocks of 8 waves
  gemm_f16_wmma<<<gemmBlocks, 256, 0, stream>>>(x16, wq16, nullptr, nullptr, q16, M, N, K);
  gemm_f16_wmma<<<gemmBlocks, 256, 0, stream>>>(x16, wk16, nullptr, nullptr, k16, M, N, K);
  gemm_f16_wmma<<<gemmBlocks, 256, 0, stream>>>(x16, wv16, nullptr, nullptr, v16, M, N, K);

  // fused logits + ALiBi + causal mask + softmax -> weights (output #2)
  attn_softmax<<<cB * cH * (cT / 16), 256, 0, stream>>>(q16, k16, wts);

  // att = diag(weights) * v
  diag_scale<<<(int)(BT_C / 256), 256, 0, stream>>>(wts, v16, att16);

  // out = att @ Wp^T + bp (f32 output #1)
  gemm_f16_wmma<<<gemmBlocks, 256, 0, stream>>>(att16, wp16, bp, out, nullptr, M, N, K);
}
